// Aggregate_61177514164583
// MI455X (gfx1250) — matrix operand order repaired and checked by reading, not compiled, not run on hardware
//
#include <hip/hip_runtime.h>
#include <hip/hip_bf16.h>

typedef __attribute__((ext_vector_type(16))) _Float16 v16h;
typedef __attribute__((ext_vector_type(8)))  float    v8f;

#define N_NODES 50000
#define HC 128           // HEADS*OUT_CH
#define NEG_SLOPE 0.2f

// Order-preserving float<->uint mapping for atomic max on floats.
__device__ __forceinline__ unsigned fenc(float f) {
  unsigned b = __float_as_uint(f);
  return (b & 0x80000000u) ? ~b : (b | 0x80000000u);
}
__device__ __forceinline__ float fdec(unsigned u) {
  unsigned b = (u & 0x80000000u) ? (u & 0x7FFFFFFFu) : ~u;
  return __uint_as_float(b);
}

// out <- bias (scatter target), maxenc <- enc(-inf), denom <- 0
__global__ __launch_bounds__(256) void gat_init(
    float* __restrict__ out, const float* __restrict__ bias,
    unsigned* __restrict__ maxenc, float* __restrict__ denom, int n_nodes) {
  int tid = blockIdx.x * blockDim.x + threadIdx.x;
  if (tid < n_nodes * HC) out[tid] = bias[tid & (HC - 1)];
  if (tid < n_nodes * 4) { maxenc[tid] = 0x007FFFFFu; denom[tid] = 0.0f; }
}

// Projections x_l = x@Wl.T + bl, x_r = x@Wr.T + br via WMMA f16->f32.
// One wave per 16x16 output tile, computing both Wl and Wr results
// (shared A fragment). Grid is exact: 3125 row-tiles * 8 col-tiles waves.
__global__ __launch_bounds__(256) void gat_proj_wmma(
    const float* __restrict__ x,
    const float* __restrict__ Wl, const float* __restrict__ bl,
    const float* __restrict__ Wr, const float* __restrict__ br,
    float* __restrict__ xl, float* __restrict__ xr) {
  const int wid    = blockIdx.x * (blockDim.x >> 5) + (threadIdx.x >> 5);
  const int tile_m = wid >> 3;          // 0..3124
  const int tile_n = wid & 7;           // 0..7
  const int lane   = threadIdx.x & 31;
  const int l15    = lane & 15;
  const int hi     = lane >> 4;         // 0 or 1

  const float* arow  = x  + (tile_m * 16 + l15) * HC;  // A row (M = lane&15)
  const float* wlrow = Wl + (tile_n * 16 + l15) * HC;  // B col = W row
  const float* wrrow = Wr + (tile_n * 16 + l15) * HC;

  v8f accl = {};
  v8f accr = {};
  for (int k0 = 0; k0 < HC; k0 += 32) {
    v16h a, b_l, b_r;
    // A 16x32 f16 layout: half e -> K = 16*hi + e (contiguous)
    const int ab = k0 + hi * 16;
#pragma unroll
    for (int e = 0; e < 8; ++e) {
      a[e]     = (_Float16)arow[ab + e];
      a[e + 8] = (_Float16)arow[ab + 8 + e];
    }
    // B 32x16 f16 layout: half e -> K = 16*hi + e (contiguous)
    const int bb = k0 + hi * 16;
#pragma unroll
    for (int e = 0; e < 16; ++e) {
      b_l[e] = (_Float16)wlrow[bb + e];
      b_r[e] = (_Float16)wrrow[bb + e];
    }
    accl = __builtin_amdgcn_wmma_f32_16x16x32_f16(
        false, a, false, b_l, (short)0, accl, false, false);
    accr = __builtin_amdgcn_wmma_f32_16x16x32_f16(
        false, a, false, b_r, (short)0, accr, false, false);
  }
  // D layout: VGPR r -> (M = r + 8*hi, N = lane&15)
  const int ncol  = tile_n * 16 + l15;
  const float bla = bl[ncol];
  const float bra = br[ncol];
#pragma unroll
  for (int r = 0; r < 8; ++r) {
    const int orow = tile_m * 16 + r + hi * 8;
    xl[orow * HC + ncol] = accl[r] + bla;
    xr[orow * HC + ncol] = accr[r] + bra;
  }
}

// Pass 1 over edges: alpha[e][h] = sum_c lrelu(xl[tgt]+xr[src]) * att; atomic max per tgt.
// One wave per edge; lane c covers channel h*32+c for each head h.
__global__ __launch_bounds__(256) void gat_edge_alpha(
    const long long* __restrict__ ei, const float* __restrict__ xl,
    const float* __restrict__ xr, const float* __restrict__ att,
    float* __restrict__ alpha, unsigned* __restrict__ maxenc,
    int n_edges, int n_nodes) {
  const int e = blockIdx.x * (blockDim.x >> 5) + (threadIdx.x >> 5);
  if (e >= n_edges + n_nodes) return;
  const int lane = threadIdx.x & 31;
  int src, tgt;
  if (e < n_edges) { src = (int)ei[e]; tgt = (int)ei[n_edges + e]; }
  else             { src = tgt = e - n_edges; }

  const float* pl = xl + (long)tgt * HC;
  const float* pr = xr + (long)src * HC;
  float s[4];
#pragma unroll
  for (int h = 0; h < 4; ++h) {
    const int c = h * 32 + lane;
    float v = pl[c] + pr[c];
    v = (v > 0.0f) ? v : v * NEG_SLOPE;
    s[h] = v * att[c];
  }
#pragma unroll
  for (int off = 16; off; off >>= 1) {
#pragma unroll
    for (int h = 0; h < 4; ++h) s[h] += __shfl_xor(s[h], off, 32);
  }
  if (lane < 4) {
    alpha[(long)e * 4 + lane] = s[lane];
    atomicMax(&maxenc[tgt * 4 + lane], fenc(s[lane]));
  }
}

// Pass 2: w = exp(alpha - max[tgt]); denom[tgt] += w. One thread per (edge,head).
__global__ __launch_bounds__(256) void gat_edge_exp(
    const long long* __restrict__ ei, float* __restrict__ alpha,
    const unsigned* __restrict__ maxenc, float* __restrict__ denom,
    int n_edges, int n_nodes) {
  const long tid = (long)blockIdx.x * blockDim.x + threadIdx.x;
  const long total = (long)(n_edges + n_nodes) * 4;
  if (tid >= total) return;
  const int e = (int)(tid >> 2);
  const int h = (int)(tid & 3);
  int tgt = (e < n_edges) ? (int)ei[n_edges + e] : (e - n_edges);
  const float m = fdec(maxenc[tgt * 4 + h]);
  const float w = __expf(alpha[tid] - m);
  alpha[tid] = w;
  atomicAdd(&denom[tgt * 4 + h], w);
}

// Pass 3: out[tgt] += xr[src] * (w / denom[tgt]). One wave per edge.
__global__ __launch_bounds__(256) void gat_edge_scatter(
    const long long* __restrict__ ei, const float* __restrict__ xr,
    const float* __restrict__ alpha, const float* __restrict__ denom,
    float* __restrict__ out, int n_edges, int n_nodes) {
  const int e = blockIdx.x * (blockDim.x >> 5) + (threadIdx.x >> 5);
  if (e >= n_edges + n_nodes) return;
  const int lane = threadIdx.x & 31;
  int src, tgt;
  if (e < n_edges) { src = (int)ei[e]; tgt = (int)ei[n_edges + e]; }
  else             { src = tgt = e - n_edges; }

  float coeff[4];
#pragma unroll
  for (int h = 0; h < 4; ++h)
    coeff[h] = alpha[(long)e * 4 + h] / denom[tgt * 4 + h];

  const float* pr = xr + (long)src * HC;
  float* po = out + (long)tgt * HC;
#pragma unroll
  for (int h = 0; h < 4; ++h) {
    const int c = h * 32 + lane;
    atomicAdd(&po[c], pr[c] * coeff[h]);
  }
}

extern "C" void kernel_launch(void* const* d_in, const int* in_sizes, int n_in,
                              void* d_out, int out_size, void* d_ws, size_t ws_size,
                              hipStream_t stream) {
  const float*     x    = (const float*)d_in[0];
  const long long* ei   = (const long long*)d_in[1];  // int64 [2, E]
  const float*     Wl   = (const float*)d_in[2];
  const float*     bl   = (const float*)d_in[3];
  const float*     Wr   = (const float*)d_in[4];
  const float*     br   = (const float*)d_in[5];
  const float*     att  = (const float*)d_in[6];
  const float*     bias = (const float*)d_in[7];
  float*           out  = (float*)d_out;

  const int N = in_sizes[0] / HC;      // 50000
  const int E = in_sizes[1] / 2;       // 800000
  const int Ep = E + N;                // with self-loops

  char* ws = (char*)d_ws;
  float*    xl     = (float*)(ws);
  float*    xr     = (float*)(ws + (size_t)N * HC * 4);
  float*    alpha  = (float*)(ws + (size_t)2 * N * HC * 4);
  unsigned* maxenc = (unsigned*)(ws + (size_t)2 * N * HC * 4 + (size_t)Ep * 4 * 4);
  float*    denom  = (float*)(ws + (size_t)2 * N * HC * 4 + (size_t)Ep * 4 * 4 +
                              (size_t)N * 4 * 4);

  // init: out<-bias, maxenc<-enc(-inf), denom<-0
  {
    const long t = (long)N * HC;
    gat_init<<<(unsigned)((t + 255) / 256), 256, 0, stream>>>(out, bias, maxenc,
                                                              denom, N);
  }
  // projections via WMMA: (N/16)*8 waves, 8 waves per block
  {
    const int waves = (N / 16) * 8;
    gat_proj_wmma<<<waves / 8, 256, 0, stream>>>(x, Wl, bl, Wr, br, xl, xr);
  }
  // edge pass 1: logits + segment max
  {
    const int blocks = (Ep + 7) / 8;
    gat_edge_alpha<<<blocks, 256, 0, stream>>>(ei, xl, xr, att, alpha, maxenc,
                                               E, N);
  }
  // edge pass 2: exp + segment sum (denominator)
  {
    const long t = (long)Ep * 4;
    gat_edge_exp<<<(unsigned)((t + 255) / 256), 256, 0, stream>>>(ei, alpha,
                                                                  maxenc, denom,
                                                                  E, N);
  }
  // edge pass 3: weighted scatter-add into out (pre-initialized with bias)
  {
    const int blocks = (Ep + 7) / 8;
    gat_edge_scatter<<<blocks, 256, 0, stream>>>(ei, xr, alpha, denom, out, E, N);
  }
}
